// GraphAttentionLayer_38972533244581
// MI455X (gfx1250) — matrix operand order repaired and checked by reading, not compiled, not run on hardware
//
#include <hip/hip_runtime.h>
#include <hip/hip_fp16.h>

typedef __attribute__((ext_vector_type(16))) _Float16 v16h;
typedef __attribute__((ext_vector_type(8)))  float    v8f;
typedef __attribute__((ext_vector_type(2)))  int      v2i;

#define AS3 __attribute__((address_space(3)))

constexpr int   NN     = 4096;   // nodes
constexpr int   IN_F   = 256;
constexpr int   OUT_F  = 64;
constexpr int   HEADS  = 8;
constexpr float ALPHA  = 0.2f;

// Fragment layouts used here (wave32; A/B use contiguous K, C/D as in ISA 7.12.2):
//  A (16x32 f16): lane = grp*16+lr covers row lr;  element e -> K = e + 16*grp
//  B (32x16 f16): lane = grp*16+lr covers col lr;  element e -> K = e + 16*grp
//  C/D (16x16 f32): vgpr r -> M = r + 8*grp, N = lr

__device__ __forceinline__ int a_K(int e, int grp) { return e + grp * 16; }

// --- async copy helpers (CDNA5 async-to-LDS; plain LDS stores as fallback) ---
__device__ __forceinline__ void adj_tile_issue(const int* gsrc, int* ldst) {
#if __has_builtin(__builtin_amdgcn_global_load_async_to_lds_b64)
    __builtin_amdgcn_global_load_async_to_lds_b64((v2i*)gsrc, (AS3 v2i*)ldst, 0, 0);
#else
    ldst[0] = gsrc[0];
    ldst[1] = gsrc[1];
#endif
}

template <int NOut>
__device__ __forceinline__ void async_wait() {
#if __has_builtin(__builtin_amdgcn_global_load_async_to_lds_b64)
#if __has_builtin(__builtin_amdgcn_s_wait_asynccnt)
    __builtin_amdgcn_s_wait_asynccnt(NOut);
#else
    asm volatile("s_wait_asynccnt %0" ::"i"(NOut) : "memory");
#endif
#endif
}

// ---------------------------------------------------------------------------
// Kernel 1a: swizzle X (f32) into fragment-major f16 A-tiles.
// Xs layout: ((rt*8 + kt)*32 + lane)*16 + e ; rt=row-tile(16), kt=K-tile(32)
// ---------------------------------------------------------------------------
__global__ void swizzleX(const float* __restrict__ X, _Float16* __restrict__ Xs) {
    int t = blockIdx.x * blockDim.x + threadIdx.x;        // NN*IN_F threads
    int e = t & 15, lane = (t >> 4) & 31, kt = (t >> 9) & 7, rt = t >> 12;
    int grp = lane >> 4, lr = lane & 15;
    int row = rt * 16 + lr;
    int col = kt * 32 + a_K(e, grp);
    Xs[t] = (_Float16)X[row * IN_F + col];
}

// ---------------------------------------------------------------------------
// Kernel 1b: swizzle W (f32) into fragment-major f16 B-tiles.
// Ws layout: ((((h*8)+kt)*4 + c)*32 + lane)*16 + e
// ---------------------------------------------------------------------------
__global__ void swizzleW(const float* __restrict__ W, _Float16* __restrict__ Ws) {
    int t = blockIdx.x * blockDim.x + threadIdx.x;        // HEADS*IN_F*OUT_F threads
    int e = t & 15, lane = (t >> 4) & 31, c = (t >> 9) & 3, kt = (t >> 11) & 7, h = t >> 14;
    int grp = lane >> 4, lr = lane & 15;
    int K = e + grp * 16;
    Ws[t] = (_Float16)W[(h * IN_F + kt * 32 + K) * OUT_F + c * 16 + lr];
}

// ---------------------------------------------------------------------------
// Kernel 2: H[h] = X @ W[h]  (f16 WMMA, f32 accum). A/B frags are single
// v16h (2x global_load_b128) loads from the swizzled buffers. Epilogue
// scatters H directly into B-operand layout for the attention matmul:
// Hs layout: ((((h*128)+mt)*4 + c)*32 + lane)*16 + e   (mt = m-tile of 32)
// ---------------------------------------------------------------------------
__global__ void __launch_bounds__(32)
gemm_H(const _Float16* __restrict__ Xs, const _Float16* __restrict__ Ws,
       _Float16* __restrict__ Hs) {
    const int lane = threadIdx.x;
    const int grp  = lane >> 4, lr = lane & 15;
    const int rt   = blockIdx.x;          // 16-row tile
    const int h    = blockIdx.y;

    v8f acc[4] = {};
#pragma unroll
    for (int kt = 0; kt < 8; ++kt) {
        v16h afrag = *(const v16h*)(Xs + ((rt * 8 + kt) * 32 + lane) * 16);
#pragma unroll
        for (int c = 0; c < 4; ++c) {
            v16h bfrag = *(const v16h*)(Ws + ((((h * 8) + kt) * 4 + c) * 32 + lane) * 16);
            acc[c] = __builtin_amdgcn_wmma_f32_16x16x32_f16(
                false, afrag, false, bfrag, (short)0, acc[c], false, false);
        }
    }
    const int rowBase = rt * 16;
#pragma unroll
    for (int c = 0; c < 4; ++c)
#pragma unroll
        for (int r = 0; r < 8; ++r) {
            int m    = rowBase + r + grp * 8;
            int mt   = m >> 5, Krel = m & 31;
            int laneB = ((Krel >> 4) << 4) + lr;
            int eB    = Krel & 15;
            Hs[((((h * 128) + mt) * 4 + c) * 32 + laneB) * 16 + eB] = (_Float16)acc[c][r];
        }
}

// ---------------------------------------------------------------------------
// Kernel 3: f1/f2 logits from the swizzled H (L2-resident).
// ---------------------------------------------------------------------------
__global__ void attn_logits(const _Float16* __restrict__ Hs,
                            const float* __restrict__ a,   // [HEADS, 2*OUT_F]
                            float* __restrict__ f1, float* __restrict__ f2) {
    int t = blockIdx.x * blockDim.x + threadIdx.x;     // over HEADS*NN
    int h = t / NN, n = t % NN;
    int mt = n >> 5, Krel = n & 31;
    int grpB = Krel >> 4, eB = Krel & 15;
    const _Float16* base = Hs + (size_t)(((h * 128) + mt) * 4) * 32 * 16;
    float s1 = 0.f, s2 = 0.f;
#pragma unroll 8
    for (int f = 0; f < OUT_F; ++f) {
        int c = f >> 4, lr = f & 15;
        float hv = (float)base[(c * 32 + grpB * 16 + lr) * 16 + eB];
        s1 += hv * a[h * 2 * OUT_F + f];
        s2 += hv * a[h * 2 * OUT_F + OUT_F + f];
    }
    f1[h * NN + n] = s1;
    f2[h * NN + n] = s2;
}

// ---------------------------------------------------------------------------
// Kernel 4: masked-softmax row stats, all 8 heads per adj read.
// ---------------------------------------------------------------------------
__global__ void __launch_bounds__(256)
row_stats(const int* __restrict__ adj,
          const float* __restrict__ f1, const float* __restrict__ f2,
          float* __restrict__ rmax, float* __restrict__ rsum) {
    const int lane = threadIdx.x & 31;
    const int wave = threadIdx.x >> 5;
    const int n    = blockIdx.x * 8 + wave;
    const float NEG = -__builtin_inff();

    float mx[HEADS], sm[HEADS], f1v[HEADS];
#pragma unroll
    for (int h = 0; h < HEADS; ++h) {
        mx[h] = NEG; sm[h] = 0.f; f1v[h] = f1[h * NN + n];
    }
    for (int m = lane; m < NN; m += 32) {
        int ad = adj[n * NN + m];
        if (ad != 0) {
#pragma unroll
            for (int h = 0; h < HEADS; ++h) {
                float x = f1v[h] + f2[h * NN + m];
                float e = (x > 0.f) ? x : ALPHA * x;
                if (e > mx[h]) {
                    sm[h] = sm[h] * __expf(mx[h] - e) + 1.f;
                    mx[h] = e;
                } else {
                    sm[h] += __expf(e - mx[h]);
                }
            }
        }
    }
    for (int off = 16; off > 0; off >>= 1) {
#pragma unroll
        for (int h = 0; h < HEADS; ++h) {
            float omx = __shfl_xor(mx[h], off, 32);
            float osm = __shfl_xor(sm[h], off, 32);
            float nm  = fmaxf(mx[h], omx);
            float pa  = (mx[h] == NEG) ? 0.f : sm[h] * __expf(mx[h] - nm);
            float pb  = (omx   == NEG) ? 0.f : osm   * __expf(omx   - nm);
            mx[h] = nm; sm[h] = pa + pb;
        }
    }
    if (lane == 0) {
#pragma unroll
        for (int h = 0; h < HEADS; ++h) {
            rmax[h * NN + n] = mx[h];
            rsum[h * NN + n] = sm[h];
        }
    }
}

// ---------------------------------------------------------------------------
// Kernel 5: H_out[h] = P @ H[h], exact two-pass softmax. 8 waves = 8 heads
// per block. adj tiles stream through a DOUBLE-BUFFERED async-to-LDS
// pipeline: issue tile t+1, s_wait_asynccnt 1 (completes tile t in-order,
// leaves t+1 in flight behind the WMMA work), barrier, compute.
// ---------------------------------------------------------------------------
__global__ void __launch_bounds__(256)
attn_matmul(const int* __restrict__ adj,
            const float* __restrict__ f1, const float* __restrict__ f2,
            const float* __restrict__ rmax, const float* __restrict__ rsum,
            const _Float16* __restrict__ Hs,
            float* __restrict__ Hhead) {        // [HEADS, NN, OUT_F]
    __shared__ int adj_s[2][16 * 32];

    const int tid     = threadIdx.x;
    const int lane    = tid & 31;
    const int h       = tid >> 5;               // wave index = head
    const int grp     = lane >> 4, lr = lane & 15;
    const int rowBase = blockIdx.x * 16;
    const int myrow   = rowBase + lr;

    const float rm    = rmax[h * NN + myrow];
    const float rsinv = 1.f / rsum[h * NN + myrow];  // >=1 (self loop)
    const float f1v   = f1[h * NN + myrow];

    // this thread's 8B slice of every 16x32 adj tile
    const int rowOff  = (rowBase + (tid >> 4)) * NN + (tid & 15) * 2;
    const int ldsOff  = tid * 2;

    adj_tile_issue(adj + rowOff, &adj_s[0][ldsOff]);           // preload tile 0

    v8f acc[4] = {};
    for (int m0 = 0; m0 < NN; m0 += 32) {
        const int mt  = m0 >> 5;
        const int buf = mt & 1;
        if (m0 + 32 < NN) {
            __syncthreads();                    // buf^1 consumers (iter t-1) done
            adj_tile_issue(adj + rowOff + m0 + 32, &adj_s[buf ^ 1][ldsOff]);
            if (m0 + 64 < NN) __builtin_prefetch(adj + rowOff + m0 + 64, 0, 1);
            async_wait<1>();                    // tile t complete, t+1 in flight
        } else {
            async_wait<0>();                    // drain final tile
        }
        __syncthreads();                        // tile t visible block-wide

        // P fragment (A-operand, f16): p = exp(lrelu(f1+f2) - rmax) / rsum
        v16h pfrag;
#pragma unroll
        for (int e = 0; e < 16; ++e) {
            int K = a_K(e, grp);
            float p = 0.f;
            if (adj_s[buf][lr * 32 + K] != 0) {
                float x  = f1v + f2[h * NN + m0 + K];
                float ee = (x > 0.f) ? x : ALPHA * x;
                p = __expf(ee - rm) * rsinv;
            }
            pfrag[e] = (_Float16)p;
        }
        // B fragments: single v16h load per col-tile from swizzled H
#pragma unroll
        for (int c = 0; c < 4; ++c) {
            v16h bfrag = *(const v16h*)(Hs + ((((h * 128) + mt) * 4 + c) * 32 + lane) * 16);
            acc[c] = __builtin_amdgcn_wmma_f32_16x16x32_f16(
                false, pfrag, false, bfrag, (short)0, acc[c], false, false);
        }
    }
#pragma unroll
    for (int c = 0; c < 4; ++c)
#pragma unroll
        for (int r = 0; r < 8; ++r) {
            int M = r + grp * 8;
            Hhead[(h * NN + rowBase + M) * OUT_F + c * 16 + lr] = acc[c][r];
        }
}

// ---------------------------------------------------------------------------
// Kernel 6: mean over heads -> d_out (full overwrite)
// ---------------------------------------------------------------------------
__global__ void head_mean(const float* __restrict__ Hhead, float* __restrict__ out) {
    int t = blockIdx.x * blockDim.x + threadIdx.x;   // over NN*OUT_F
    float s = 0.f;
#pragma unroll
    for (int h = 0; h < HEADS; ++h) s += Hhead[h * NN * OUT_F + t];
    out[t] = s * (1.0f / HEADS);
}

// ---------------------------------------------------------------------------
extern "C" void kernel_launch(void* const* d_in, const int* in_sizes, int n_in,
                              void* d_out, int out_size, void* d_ws, size_t ws_size,
                              hipStream_t stream) {
    const float* X   = (const float*)d_in[0];   // [NN, IN_F]
    const int*   adj = (const int*)  d_in[1];   // [NN, NN]
    const float* W   = (const float*)d_in[2];   // [HEADS, IN_F, OUT_F]
    const float* a   = (const float*)d_in[3];   // [HEADS, 2*OUT_F, 1]
    float*       out = (float*)d_out;

    char* p = (char*)d_ws;
    _Float16* Xs    = (_Float16*)p;  p += sizeof(_Float16) * NN * IN_F;
    _Float16* Ws    = (_Float16*)p;  p += sizeof(_Float16) * HEADS * IN_F * OUT_F;
    _Float16* Hs    = (_Float16*)p;  p += sizeof(_Float16) * HEADS * NN * OUT_F;
    float*    f1    = (float*)p;     p += sizeof(float) * HEADS * NN;
    float*    f2    = (float*)p;     p += sizeof(float) * HEADS * NN;
    float*    rmax  = (float*)p;     p += sizeof(float) * HEADS * NN;
    float*    rsum  = (float*)p;     p += sizeof(float) * HEADS * NN;
    float*    Hhead = (float*)p;     p += sizeof(float) * HEADS * NN * OUT_F;
    (void)ws_size; (void)in_sizes; (void)n_in; (void)out_size;

    swizzleX<<<(NN * IN_F) / 256, 256, 0, stream>>>(X, Xs);
    swizzleW<<<(HEADS * IN_F * OUT_F) / 256, 256, 0, stream>>>(W, Ws);

    gemm_H<<<dim3(NN / 16, HEADS), 32, 0, stream>>>(Xs, Ws, Hs);

    attn_logits<<<(HEADS * NN) / 256, 256, 0, stream>>>(Hs, a, f1, f2);

    row_stats<<<NN / 8, 256, 0, stream>>>(adj, f1, f2, rmax, rsum);

    attn_matmul<<<NN / 16, 256, 0, stream>>>(adj, f1, f2, rmax, rsum, Hs, Hhead);

    head_mean<<<(NN * OUT_F) / 256, 256, 0, stream>>>(Hhead, out);
}
